// MultiHeadAttention_unboxed_7146825581205
// MI455X (gfx1250) — matrix operand order repaired and checked by reading, not compiled, not run on hardware
//
#include <hip/hip_runtime.h>

// ---------------------------------------------------------------------------
// MI455X (gfx1250) multi-head attention block, bf16 WMMA pipeline.
// All GEMMs via v_wmma_f32_16x16x32_bf16; softmax/LayerNorm stats in f32.
// ---------------------------------------------------------------------------

typedef __attribute__((ext_vector_type(16))) __bf16 v16bf;
typedef __attribute__((ext_vector_type(8)))  float  v8f;

#define WMMA_BF16(A, B, C) \
  __builtin_amdgcn_wmma_f32_16x16x32_bf16(false, (A), false, (B), (short)0, (C), false, false)

static __device__ __forceinline__ unsigned short f2bf(float f) {
  unsigned int u = __float_as_uint(f);
  u += 0x7FFFu + ((u >> 16) & 1u);            // round-to-nearest-even
  return (unsigned short)(u >> 16);
}

// A-matrix (16x32 bf16, M x K) fragment K-base for VGPR v, lane-half  (ISA 7.12.2)
static __device__ __forceinline__ int a_kbase(int v, int half) {
  return (v < 4 ? 2 * v : 16 + 2 * (v - 4)) + 8 * half;
}
// B-matrix (32x16 bf16, K x N) fragment K-base for VGPR v, lane-half
static __device__ __forceinline__ int b_kbase(int v, int half) {
  return (v < 4 ? 2 * v : 16 + 2 * (v - 4)) + 8 * half;
}

// ---------------------------------------------------------------------------
// Pack kernels: plain row-major (f32 or bf16) -> fragment-major bf16 tiles.
// One thread per packed dword (2 bf16). Tile = 512 elems = 256 dwords.
// ---------------------------------------------------------------------------

// A-fragments: tiles over (R/16) x (C/32), row-major tile grid.
__global__ void pack_a_kernel(const void* __restrict__ src, unsigned short* __restrict__ dst,
                              int R, int C, int NB, int srcIsF32) {
  long long idx = (long long)blockIdx.x * blockDim.x + threadIdx.x;
  long long per = (long long)R * C / 2;
  if (idx >= per * NB) return;
  int batch = (int)(idx / per);
  int r0 = (int)(idx - (long long)batch * per);
  int tile = r0 >> 8, r = r0 & 255;
  int lane = r >> 3, v = r & 7;
  int tcols = C / 32;
  int ti = tile / tcols, tj = tile - ti * tcols;
  int m = 16 * ti + (lane & 15);
  int k = 32 * tj + a_kbase(v, lane >> 4);
  long long sb = (long long)batch * R * C + (long long)m * C + k;
  unsigned int lo, hi;
  if (srcIsF32) {
    const float* s = (const float*)src;
    lo = f2bf(s[sb]); hi = f2bf(s[sb + 1]);
  } else {
    const unsigned short* s = (const unsigned short*)src;
    lo = s[sb]; hi = s[sb + 1];
  }
  ((unsigned int*)dst)[idx] = lo | (hi << 16);
}

// B-fragments of src as-is (src is K_total x N_total): tiles (R/32) x (C/16).
__global__ void pack_b_kernel(const void* __restrict__ src, unsigned short* __restrict__ dst,
                              int R, int C, int NB, int srcIsF32) {
  long long idx = (long long)blockIdx.x * blockDim.x + threadIdx.x;
  long long per = (long long)R * C / 2;
  if (idx >= per * NB) return;
  int batch = (int)(idx / per);
  int r0 = (int)(idx - (long long)batch * per);
  int tile = r0 >> 8, r = r0 & 255;
  int lane = r >> 3, v = r & 7;
  int tcols = C / 16;
  int ti = tile / tcols, tj = tile - ti * tcols;
  int n = 16 * tj + (lane & 15);
  int k = 32 * ti + b_kbase(v, lane >> 4);
  long long sb = (long long)batch * R * C + (long long)k * C + n;
  unsigned int lo, hi;
  if (srcIsF32) {
    const float* s = (const float*)src;
    lo = f2bf(s[sb]); hi = f2bf(s[sb + C]);
  } else {
    const unsigned short* s = (const unsigned short*)src;
    lo = s[sb]; hi = s[sb + C];
  }
  ((unsigned int*)dst)[idx] = lo | (hi << 16);
}

// B-fragments of src^T (src is R x C row-major; B = src^T so K_total=C, N_total=R):
// tiles (C/32) x (R/16). The two K-consecutive elements are contiguous in src.
__global__ void pack_bt_kernel(const void* __restrict__ src, unsigned short* __restrict__ dst,
                               int R, int C, int NB, int srcIsF32) {
  long long idx = (long long)blockIdx.x * blockDim.x + threadIdx.x;
  long long per = (long long)R * C / 2;
  if (idx >= per * NB) return;
  int batch = (int)(idx / per);
  int r0 = (int)(idx - (long long)batch * per);
  int tile = r0 >> 8, r = r0 & 255;
  int lane = r >> 3, v = r & 7;
  int tcols = R / 16;
  int ti = tile / tcols, tj = tile - ti * tcols;
  int n = 16 * tj + (lane & 15);                 // src row
  int k = 32 * ti + b_kbase(v, lane >> 4);       // src col
  long long sb = (long long)batch * R * C + (long long)n * C + k;
  unsigned int lo, hi;
  if (srcIsF32) {
    const float* s = (const float*)src;
    lo = f2bf(s[sb]); hi = f2bf(s[sb + 1]);
  } else {
    const unsigned short* s = (const unsigned short*)src;
    lo = s[sb]; hi = s[sb + 1];
  }
  ((unsigned int*)dst)[idx] = lo | (hi << 16);
}

// ---------------------------------------------------------------------------
// QKV projection GEMM: per (b,h): [S x D] @ [D x 64] -> plain bf16 [S x 64].
// 4 waves/block, each wave one 16(M) x 64(N) strip; K-loop in steps of 32.
// ---------------------------------------------------------------------------
__global__ __launch_bounds__(128) void gemm_proj_kernel(
    const unsigned short* __restrict__ Apack, const unsigned short* __restrict__ Bpack,
    unsigned short* __restrict__ out, int S, int D, int H) {
  int wid = threadIdx.x >> 5, lane = threadIdx.x & 31;
  int half = lane >> 4, nl = lane & 15;
  int ti = blockIdx.x * 4 + wid;
  int bh = blockIdx.y;
  int b = bh / H, h = bh - b * H;
  int KT = D / 32;
  const unsigned short* Ab = Apack + (long long)b * S * D + (long long)ti * KT * 512;
  const unsigned short* Bb = Bpack + (long long)h * D * 64;
  v8f z = {0.f, 0.f, 0.f, 0.f, 0.f, 0.f, 0.f, 0.f};
  v8f a0 = z, a1 = z, a2 = z, a3 = z;
  for (int kk = 0; kk < KT; ++kk) {
    v16bf af = *(const v16bf*)(Ab + (long long)kk * 512 + lane * 16);
    const unsigned short* bp = Bb + (long long)kk * 4 * 512 + lane * 16;
    v16bf b0 = *(const v16bf*)(bp);
    v16bf b1 = *(const v16bf*)(bp + 512);
    v16bf b2 = *(const v16bf*)(bp + 1024);
    v16bf b3 = *(const v16bf*)(bp + 1536);
    a0 = WMMA_BF16(af, b0, a0);
    a1 = WMMA_BF16(af, b1, a1);
    a2 = WMMA_BF16(af, b2, a2);
    a3 = WMMA_BF16(af, b3, a3);
  }
  unsigned short* ob = out + ((long long)bh * S + ti * 16) * 64;
#pragma unroll
  for (int v = 0; v < 8; ++v) {
    int m = v + 8 * half;
    ob[m * 64 + nl + 0]  = f2bf(a0[v]);
    ob[m * 64 + nl + 16] = f2bf(a1[v]);
    ob[m * 64 + nl + 32] = f2bf(a2[v]);
    ob[m * 64 + nl + 48] = f2bf(a3[v]);
  }
}

// ---------------------------------------------------------------------------
// Flash attention: one wave per (b,h,16-query block). Online softmax in LDS.
// ---------------------------------------------------------------------------
__global__ __launch_bounds__(32) void flash_attn_kernel(
    const unsigned short* __restrict__ qhA, const unsigned short* __restrict__ khBT,
    const unsigned short* __restrict__ vhB, const int* __restrict__ mask,
    unsigned short* __restrict__ concat, int S, int H) {
  const float invT = 0.125f;           // 1/sqrt(DK=64)
  const float NEGI = -1.0e9f;
  int lane = threadIdx.x;
  int half = lane >> 4, nl = lane & 15;
  int qt = blockIdx.x, h = blockIdx.y, b = blockIdx.z;
  long long hb = (long long)(b * H + h) * S * 64;
  const unsigned short* Qa = qhA + hb;   // A-frags: tiles (S/16) x 2
  const unsigned short* Kb = khBT + hb;  // B-frags of kh^T: tiles 2 x (S/16)
  const unsigned short* Vb = vhB + hb;   // B-frags of vh: tiles (S/32) x 4
  int NT = S / 16;

  v16bf qa0 = *(const v16bf*)(Qa + (long long)(qt * 2 + 0) * 512 + lane * 16);
  v16bf qa1 = *(const v16bf*)(Qa + (long long)(qt * 2 + 1) * 512 + lane * 16);

  __shared__ float stage[16 * 32];
  __shared__ unsigned short Parr[16 * 32];
  __shared__ float mI[16], lI[16], aI[16];
  if (lane < 16) { mI[lane] = -1.0e30f; lI[lane] = 0.0f; }
  __syncthreads();

  v8f z = {0.f, 0.f, 0.f, 0.f, 0.f, 0.f, 0.f, 0.f};
  v8f o0 = z, o1 = z, o2 = z, o3 = z;

  for (int j = 0; j < S / 32; ++j) {
#pragma unroll
    for (int jj = 0; jj < 2; ++jj) {
      int kt = j * 2 + jj;
      v8f s = z;
      s = WMMA_BF16(qa0, *(const v16bf*)(Kb + (long long)(0 * NT + kt) * 512 + lane * 16), s);
      s = WMMA_BF16(qa1, *(const v16bf*)(Kb + (long long)(1 * NT + kt) * 512 + lane * 16), s);
      const int* mrow = mask + ((long long)b * S + qt * 16) * S + kt * 16 + nl;
#pragma unroll
      for (int vv = 0; vv < 8; ++vv) {
        int m = vv + 8 * half;
        float val = s[vv] * invT;
        if (mrow[(long long)m * S] == 0) val = NEGI;
        stage[m * 32 + jj * 16 + nl] = val;
      }
    }
    __syncthreads();
    if (lane < 16) {                     // lane r owns softmax row r
      float mp = mI[lane], mx = mp;
      for (int c = 0; c < 32; ++c) mx = fmaxf(mx, stage[lane * 32 + c]);
      float al = __expf(mp - mx), ps = 0.f;
      for (int c = 0; c < 32; ++c) {
        float p = __expf(stage[lane * 32 + c] - mx);
        Parr[lane * 32 + c] = f2bf(p);
        ps += p;
      }
      lI[lane] = al * lI[lane] + ps;
      mI[lane] = mx;
      aI[lane] = al;
    }
    __syncthreads();
    float av[8];
#pragma unroll
    for (int vv = 0; vv < 8; ++vv) av[vv] = aI[vv + 8 * half];
#pragma unroll
    for (int vv = 0; vv < 8; ++vv) {
      o0[vv] *= av[vv]; o1[vv] *= av[vv]; o2[vv] *= av[vv]; o3[vv] *= av[vv];
    }
    union { unsigned int u[8]; v16bf bv; } pu;   // rebuild P as A-fragment
#pragma unroll
    for (int vv = 0; vv < 8; ++vv) {
      int kb = a_kbase(vv, half);
      pu.u[vv] = *(const unsigned int*)&Parr[nl * 32 + kb];
    }
    const unsigned short* vp = Vb + (long long)(j * 4) * 512 + lane * 16;
    o0 = WMMA_BF16(pu.bv, *(const v16bf*)(vp), o0);
    o1 = WMMA_BF16(pu.bv, *(const v16bf*)(vp + 512), o1);
    o2 = WMMA_BF16(pu.bv, *(const v16bf*)(vp + 1024), o2);
    o3 = WMMA_BF16(pu.bv, *(const v16bf*)(vp + 1536), o3);
    __syncthreads();
  }

  float li[8];
#pragma unroll
  for (int vv = 0; vv < 8; ++vv) li[vv] = 1.0f / lI[vv + 8 * half];
  unsigned short* orow = concat + ((long long)b * S + qt * 16) * (H * 64) + h * 64;
#pragma unroll
  for (int vv = 0; vv < 8; ++vv) {
    int m = vv + 8 * half;
    long long rb = (long long)m * (H * 64);
    orow[rb + 0 + nl]  = f2bf(o0[vv] * li[vv]);
    orow[rb + 16 + nl] = f2bf(o1[vv] * li[vv]);
    orow[rb + 32 + nl] = f2bf(o2[vv] * li[vv]);
    orow[rb + 48 + nl] = f2bf(o3[vv] * li[vv]);
  }
}

// ---------------------------------------------------------------------------
// Output FC GEMM + residual: [M x K] @ fc_w^T -> f32 pre-LN [M x N].
// ---------------------------------------------------------------------------
__global__ __launch_bounds__(128) void gemm_fc_kernel(
    const unsigned short* __restrict__ Apack, const unsigned short* __restrict__ Bpack,
    const float* __restrict__ resid, float* __restrict__ out, int M, int N, int K) {
  int wid = threadIdx.x >> 5, lane = threadIdx.x & 31;
  int half = lane >> 4, nl = lane & 15;
  int ti = blockIdx.x * 4 + wid;     // M/16 tiles
  int ns = blockIdx.y;               // N/64 strips
  int KT = K / 32, NTB = N / 16;
  const unsigned short* Ab = Apack + (long long)ti * KT * 512;
  v8f z = {0.f, 0.f, 0.f, 0.f, 0.f, 0.f, 0.f, 0.f};
  v8f a0 = z, a1 = z, a2 = z, a3 = z;
  for (int kk = 0; kk < KT; ++kk) {
    v16bf af = *(const v16bf*)(Ab + (long long)kk * 512 + lane * 16);
    const unsigned short* bp = Bpack + ((long long)kk * NTB + ns * 4) * 512 + lane * 16;
    v16bf b0 = *(const v16bf*)(bp);
    v16bf b1 = *(const v16bf*)(bp + 512);
    v16bf b2 = *(const v16bf*)(bp + 1024);
    v16bf b3 = *(const v16bf*)(bp + 1536);
    a0 = WMMA_BF16(af, b0, a0);
    a1 = WMMA_BF16(af, b1, a1);
    a2 = WMMA_BF16(af, b2, a2);
    a3 = WMMA_BF16(af, b3, a3);
  }
#pragma unroll
  for (int v = 0; v < 8; ++v) {
    int m = v + 8 * half;
    long long rb = (long long)(ti * 16 + m) * N + ns * 64 + nl;
    out[rb + 0]  = a0[v] + resid[rb + 0];
    out[rb + 16] = a1[v] + resid[rb + 16];
    out[rb + 32] = a2[v] + resid[rb + 32];
    out[rb + 48] = a3[v] + resid[rb + 48];
  }
}

// ---------------------------------------------------------------------------
// LayerNorm: one 256-thread block per row of D=1024.
// ---------------------------------------------------------------------------
__global__ __launch_bounds__(256) void ln_kernel(
    const float* __restrict__ pre, const float* __restrict__ g,
    const float* __restrict__ bta, float* __restrict__ out, int D) {
  int row = blockIdx.x, t = threadIdx.x;
  const float* x = pre + (long long)row * D;
  float s = 0.f, s2 = 0.f;
  for (int i = t; i < D; i += 256) { float v = x[i]; s += v; s2 += v * v; }
  __shared__ float rs[256], rs2[256];
  rs[t] = s; rs2[t] = s2;
  __syncthreads();
  for (int off = 128; off > 0; off >>= 1) {
    if (t < off) { rs[t] += rs[t + off]; rs2[t] += rs2[t + off]; }
    __syncthreads();
  }
  float mu = rs[0] / (float)D;
  float var = rs2[0] / (float)D - mu * mu;
  float rstd = rsqrtf(var + 1e-6f);
  float* o = out + (long long)row * D;
  for (int i = t; i < D; i += 256) o[i] = (x[i] - mu) * rstd * g[i] + bta[i];
}

// ---------------------------------------------------------------------------
extern "C" void kernel_launch(void* const* d_in, const int* in_sizes, int n_in,
                              void* d_out, int out_size, void* d_ws, size_t ws_size,
                              hipStream_t stream) {
  const int B = 4, S = 1024, D = 1024, H = 16, DK = 64, DV = 64;
  const float* q    = (const float*)d_in[0];
  const float* k    = (const float*)d_in[1];
  const float* v    = (const float*)d_in[2];
  const int*   mask = (const int*)d_in[3];
  const float* wq   = (const float*)d_in[4];
  const float* wk   = (const float*)d_in[5];
  const float* wv   = (const float*)d_in[6];
  const float* fcw  = (const float*)d_in[7];
  const float* lng  = (const float*)d_in[8];
  const float* lnb  = (const float*)d_in[9];
  float* out = (float*)d_out;

  char* ws = (char*)d_ws;
  const size_t MB = 1ull << 20;
  // region A: input A-frag packs, reused for attention operand packs
  unsigned short* qA   = (unsigned short*)(ws + 0 * MB);   // 8MB each
  unsigned short* kA   = (unsigned short*)(ws + 8 * MB);
  unsigned short* vA   = (unsigned short*)(ws + 16 * MB);
  unsigned short* qhA  = qA;                                // reuse after proj
  unsigned short* khBT = kA;
  unsigned short* vhB  = vA;
  // region B: plain projections, reused for concat + pre-LN
  unsigned short* qh = (unsigned short*)(ws + 24 * MB);
  unsigned short* kh = (unsigned short*)(ws + 32 * MB);
  unsigned short* vh = (unsigned short*)(ws + 40 * MB);
  unsigned short* concat = qh;                              // 8MB, qh dead
  float* pre = (float*)(ws + 32 * MB);                      // 16MB, kh/vh dead
  // region C: weight B-frags, reused for fc weight pack
  unsigned short* wqB = (unsigned short*)(ws + 48 * MB);    // 2MB each
  unsigned short* wkB = (unsigned short*)(ws + 50 * MB);
  unsigned short* wvB = (unsigned short*)(ws + 52 * MB);
  unsigned short* fcBT = wqB;                               // reuse after proj
  // region D
  unsigned short* concatA = (unsigned short*)(ws + 54 * MB); // 8MB -> total 62MB

  dim3 blk(256);
  auto gridFor = [](long long elems) {
    return dim3((unsigned)((elems / 2 + 255) / 256));
  };

  // 1) pack fp32 inputs -> bf16 fragment-major
  pack_a_kernel<<<gridFor((long long)B * S * D), blk, 0, stream>>>(q, qA, S, D, B, 1);
  pack_a_kernel<<<gridFor((long long)B * S * D), blk, 0, stream>>>(k, kA, S, D, B, 1);
  pack_a_kernel<<<gridFor((long long)B * S * D), blk, 0, stream>>>(v, vA, S, D, B, 1);
  pack_b_kernel<<<gridFor((long long)H * D * DK), blk, 0, stream>>>(wq, wqB, D, DK, H, 1);
  pack_b_kernel<<<gridFor((long long)H * D * DK), blk, 0, stream>>>(wk, wkB, D, DK, H, 1);
  pack_b_kernel<<<gridFor((long long)H * D * DK), blk, 0, stream>>>(wv, wvB, D, DK, H, 1);

  // 2) QKV projections
  dim3 pgrid(S / 16 / 4, B * H);
  gemm_proj_kernel<<<pgrid, 128, 0, stream>>>(qA, wqB, qh, S, D, H);
  gemm_proj_kernel<<<pgrid, 128, 0, stream>>>(kA, wkB, kh, S, D, H);
  gemm_proj_kernel<<<pgrid, 128, 0, stream>>>(vA, wvB, vh, S, D, H);

  // 3) repack projections into attention operand layouts
  pack_a_kernel <<<gridFor((long long)B * H * S * DK), blk, 0, stream>>>(qh, qhA, S, DK, B * H, 0);
  pack_bt_kernel<<<gridFor((long long)B * H * S * DK), blk, 0, stream>>>(kh, khBT, S, DK, B * H, 0);
  pack_b_kernel <<<gridFor((long long)B * H * S * DV), blk, 0, stream>>>(vh, vhB, S, DV, B * H, 0);

  // 4) fused flash attention -> concat heads (plain bf16)
  flash_attn_kernel<<<dim3(S / 16, H, B), 32, 0, stream>>>(qhA, khBT, vhB, mask, concat, S, H);

  // 5) pack for output FC
  pack_a_kernel <<<gridFor((long long)B * S * H * DV), blk, 0, stream>>>(concat, concatA, B * S, H * DV, 1, 0);
  pack_bt_kernel<<<gridFor((long long)D * H * DV), blk, 0, stream>>>(fcw, fcBT, D, H * DV, 1, 1);

  // 6) FC + residual -> f32 pre-LN
  gemm_fc_kernel<<<dim3(B * S / 16 / 4, D / 64), 128, 0, stream>>>(concatA, fcBT, q, pre, B * S, D, H * DV);

  // 7) LayerNorm -> output
  ln_kernel<<<dim3(B * S), 256, 0, stream>>>(pre, lng, lnb, out, D);

  (void)in_sizes; (void)n_in; (void)out_size; (void)ws_size;
}
